// PureCartesianTensorProductO3_37967510897141
// MI455X (gfx1250) — matrix operand order repaired and checked by reading, not compiled, not run on hardware
//
#include <hip/hip_runtime.h>

// ---------------------------------------------------------------------------
// O(3) Cartesian tensor product, fused, WMMA (gfx1250 / CDNA5, wave32).
//
// out[n, sout, Lout, Z, m] = sum over paths/parities of
//     sum_{a,b} W[Z,a,b] * contract_spatial(x1[n,s1,L1,a,:], x2[n,s2,L2,b,:])
//
// Mapping: D(16x16) = A(16x32) x B(32x16) + C, accumulated over K=a*32+b:
//   A[row=n][k=b]  = P[n,a,b,m]   (f16, built in VALU from LDS x-tiles)
//   B[k=b][col=Z]  = W[Z,a,b]     (f16, streamed from L2-resident weights)
//   C/D            = f32 accumulators, 3^Lout components x 2 Z-tiles
// 60 path-instances grouped into 6 (sout,Lout) buckets; one wave per bucket
// -> disjoint output blocks, plain stores, no atomics, deterministic.
// ---------------------------------------------------------------------------

typedef __attribute__((ext_vector_type(16))) _Float16 v16h;
typedef __attribute__((ext_vector_type(8)))  float    v8f;

union AF { v16h v; _Float16 e[16]; };

__host__ __device__ constexpr int p3(int n) { int r = 1; for (int i = 0; i < n; ++i) r *= 3; return r; }

#define C1C2   1024     // C1*C2
#define WBLK   32768    // COUT*C1*C2
#define FEAT   832      // 2*(1+3+9)*32
#define TILE   16

__device__ __forceinline__ int blk_off(int s, int L) {
  return s * 416 + (L == 0 ? 0 : (L == 1 ? 32 : 128));
}

// One path-instance: accumulate its GEMM over K=1024 into acc[][].
template<int L1, int L2, int LOUT, int KC, int UE>
__device__ __forceinline__ void path_accum(v8f (&acc)[p3(LOUT)][2],
                                           int pathIdx, int s1, int s2, int lane,
                                           const _Float16* __restrict__ sx1,
                                           const _Float16* __restrict__ sx2,
                                           const float*    __restrict__ w)
{
  constexpr int MN   = p3(LOUT);
  constexpr int P3K  = p3(KC);
  constexpr int P3L1 = p3(L1);
  constexpr int P3L2 = p3(L2);

  const int off1 = blk_off(s1, L1);
  const int off2 = blk_off(s2, L2);
  const size_t wbase = (size_t)((pathIdx * 2 + s1) * 2 + s2) * WBLK;

  const int rowA   = lane & 15;            // A-matrix row (M) held by this lane
  const int hiHalf = (lane >= 16);
  const int baseKA = hiHalf ? 16 : 0;      // A layout: K contiguous 0-15 vs 16-31
  const int baseKB = hiHalf ? 16 : 0;      // B layout: K contiguous 0-15 vs 16-31
  const int Zl     = lane & 15;            // B column (N) held by this lane

  const _Float16* x1row = sx1 + rowA * FEAT + off1;
  const _Float16* x2row = sx2 + rowA * FEAT + off2;

  #pragma unroll 1
  for (int a = 0; a < 32; ++a) {
    // ---- B fragments: W[Z, a, b], b contiguous over 16 halves (two Z tiles) ----
    AF B0, B1;
    {
      const float* wp0 = w + wbase + (size_t)Zl * C1C2 + a * 32 + baseKB;
      const float* wp1 = wp0 + 16 * C1C2;
      #pragma unroll
      for (int h = 0; h < 16; ++h) {
        B0.e[h] = (_Float16)wp0[h];
        B1.e[h] = (_Float16)wp1[h];
      }
    }
    // ---- x1 spatial values for (rowA, a): b-invariant, hoisted to registers ----
    float xv[P3L1];
    #pragma unroll
    for (int t = 0; t < P3L1; ++t) xv[t] = (float)x1row[a * P3L1 + t];

    #pragma unroll
    for (int m = 0; m < MN; ++m) {
      // ---- A fragment: P[n, a, b(h), m] with K contiguous per lane half (b = baseKA + h) ----
      AF A;
      #pragma unroll
      for (int h = 0; h < 16; ++h) {
        const int b = baseKA + h;
        const _Float16* x2b = x2row + b * P3L2;
        float s = 0.f;
        if constexpr (!UE) {
          constexpr int NJ = p3(L2 - KC);
          const int ifree = m / NJ;
          const int jfree = m % NJ;
          #pragma unroll
          for (int c = 0; c < P3K; ++c)
            s += xv[ifree * P3K + c] * (float)x2b[jfree * P3K + c];
        } else {
          constexpr int NJE = p3(L2 - KC - 1);
          const int pp   = m % 3;
          const int rest = m / 3;
          const int jf   = rest % NJE;
          const int ifr  = rest / NJE;
          const int e1 = (pp + 1) % 3;     // eps[p, p+1, p+2] = +1
          const int e2 = (pp + 2) % 3;     // eps[p, p+2, p+1] = -1
          #pragma unroll
          for (int c = 0; c < P3K; ++c) {
            s += xv[(ifr * 3 + e1) * P3K + c] * (float)x2b[(jf * 3 + e2) * P3K + c];
            s -= xv[(ifr * 3 + e2) * P3K + c] * (float)x2b[(jf * 3 + e1) * P3K + c];
          }
        }
        A.e[h] = (_Float16)s;
      }
      // ---- D = A x B + C (fp32 accumulate), two 16-wide Z tiles ----
      acc[m][0] = __builtin_amdgcn_wmma_f32_16x16x32_f16(
          false, A.v, false, B0.v, (short)0, acc[m][0], false, false);
      acc[m][1] = __builtin_amdgcn_wmma_f32_16x16x32_f16(
          false, A.v, false, B1.v, (short)0, acc[m][1], false, false);
    }
  }
}

// One bucket = all path-instances landing in output block (sout, LOUT).
template<int LOUT>
__device__ void run_bucket(int sout, int lane, int row0,
                           const _Float16* __restrict__ sx1,
                           const _Float16* __restrict__ sx2,
                           const float*    __restrict__ w,
                           float*          __restrict__ out)
{
  constexpr int MN = p3(LOUT);
  v8f acc[MN][2];
  #pragma unroll
  for (int m = 0; m < MN; ++m) {
    acc[m][0] = {};
    acc[m][1] = {};
  }

  // Path table: (id, L1, L2, Lout, k, use_eps) in reference enumeration order.
#define RUN_PATH(ID, L1_, L2_, LO_, K_, UE_)                                   \
  if constexpr ((LO_) == LOUT) {                                               \
    for (int s1 = 0; s1 < 2; ++s1)                                             \
      for (int s2 = 0; s2 < 2; ++s2)                                           \
        if (((s1 ^ s2 ^ (UE_)) & 1) == sout)                                   \
          path_accum<L1_, L2_, LOUT, K_, UE_>(acc, ID, s1, s2, lane,           \
                                              sx1, sx2, w);                    \
  }
  RUN_PATH( 0, 0, 0, 0, 0, 0)
  RUN_PATH( 1, 0, 1, 1, 0, 0)
  RUN_PATH( 2, 0, 2, 2, 0, 0)
  RUN_PATH( 3, 1, 0, 1, 0, 0)
  RUN_PATH( 4, 1, 1, 2, 0, 0)
  RUN_PATH( 5, 1, 1, 0, 1, 0)
  RUN_PATH( 6, 1, 1, 1, 0, 1)
  RUN_PATH( 7, 1, 2, 1, 1, 0)
  RUN_PATH( 8, 1, 2, 2, 0, 1)
  RUN_PATH( 9, 2, 0, 2, 0, 0)
  RUN_PATH(10, 2, 1, 1, 1, 0)
  RUN_PATH(11, 2, 1, 2, 0, 1)
  RUN_PATH(12, 2, 2, 2, 1, 0)
  RUN_PATH(13, 2, 2, 0, 2, 0)
  RUN_PATH(14, 2, 2, 1, 1, 1)
#undef RUN_PATH

  // Store: this wave exclusively owns output block (sout, LOUT) of this tile.
  // C/D layout: VGPR r -> M = r + 8*(lane>=16), N = lane & 15.
  const int offo = blk_off(sout, LOUT);
  const int nbase = (lane < 16) ? 0 : 8;
  #pragma unroll
  for (int m = 0; m < MN; ++m)
    #pragma unroll
    for (int zt = 0; zt < 2; ++zt)
      #pragma unroll
      for (int r = 0; r < 8; ++r) {
        const int n = nbase + r;
        const int Z = (lane & 15) + 16 * zt;
        out[(size_t)(row0 + n) * FEAT + offo + Z * MN + m] = acc[m][zt][r];
      }
}

__global__ __launch_bounds__(64)
void tp_o3_wmma_kernel(const float* __restrict__ x1,
                       const float* __restrict__ x2,
                       const float* __restrict__ w,
                       float* __restrict__ out)
{
  __shared__ _Float16 sx1[TILE * FEAT];   // 26.6 KB
  __shared__ _Float16 sx2[TILE * FEAT];   // 26.6 KB

  const int row0 = blockIdx.x * TILE;

  // Cooperative load + fp32 -> f16 conversion of the 16-row x tiles.
  for (int i = threadIdx.x; i < TILE * FEAT; i += 64) {
    sx1[i] = (_Float16)x1[(size_t)row0 * FEAT + i];
    sx2[i] = (_Float16)x2[(size_t)row0 * FEAT + i];
  }
  __syncthreads();

  const int wave = threadIdx.x >> 5;
  const int lane = threadIdx.x & 31;

  // 6 buckets over (sout, Lout), paired 2 per block (one per wave):
  //   y=0: (0,L0)+(0,L2)   y=1: (1,L0)+(1,L2)   y=2: (0,L1)+(1,L1)
  int L, sout;
  if (blockIdx.y == 2) { L = 1;            sout = wave; }
  else                 { L = wave ? 2 : 0; sout = blockIdx.y; }

  if      (L == 0) run_bucket<0>(sout, lane, row0, sx1, sx2, w, out);
  else if (L == 1) run_bucket<1>(sout, lane, row0, sx1, sx2, w, out);
  else             run_bucket<2>(sout, lane, row0, sx1, sx2, w, out);
}

extern "C" void kernel_launch(void* const* d_in, const int* in_sizes, int n_in,
                              void* d_out, int out_size, void* d_ws, size_t ws_size,
                              hipStream_t stream)
{
  (void)n_in; (void)out_size; (void)d_ws; (void)ws_size;
  const float* x1 = (const float*)d_in[0];
  const float* x2 = (const float*)d_in[1];
  const float* w  = (const float*)d_in[2];
  float* out = (float*)d_out;

  const int rows = in_sizes[0] / FEAT;     // 8192
  dim3 grid(rows / TILE, 3, 1);            // 512 x 3 blocks
  tp_o3_wmma_kernel<<<grid, 64, 0, stream>>>(x1, x2, w, out);
}
